// MaskedSSMEncoder_37357625541382
// MI455X (gfx1250) — compile-verified, matrix operand order checked by reading, not run on hardware
//
#include <hip/hip_runtime.h>
#include <hip/hip_bf16.h>
#include <math.h>

// ---------------------------------------------------------------------------
// MaskedSSMEncoder for MI455X (gfx1250, wave32, WMMA + TDM).
// GEMMs: v_wmma_f32_16x16x32_bf16 (fp32 accumulate). Tile movement for the
// bf16-resident GEMMs uses the Tensor Data Mover (tensor_load_to_lds) with
// double-buffered LDS, overlapping DMA of tile k+1 with WMMA on tile k.
// Selective scan stays fp32 VALU (tiny).
// ---------------------------------------------------------------------------

typedef __bf16 bf16_t;
typedef __attribute__((ext_vector_type(16))) __bf16 v16bf;
typedef __attribute__((ext_vector_type(8)))  float  v8f;
typedef __attribute__((ext_vector_type(4)))  unsigned int u32x4;
typedef __attribute__((ext_vector_type(8)))  int i32x8;
typedef __attribute__((ext_vector_type(4)))  int i32x4;

#define D_IN     1536
#define D_MODEL  128
#define T_SEQ    12
#define BATCH    2048
#define BT       (BATCH * T_SEQ)

#define WMMA_BF16(A_, B_, C_) \
  __builtin_amdgcn_wmma_f32_16x16x32_bf16(false, (A_), false, (B_), (short)0, (C_), false, false)

__device__ __forceinline__ v8f zero_v8f() {
  v8f v;
#pragma unroll
  for (int i = 0; i < 8; ++i) v[i] = 0.0f;
  return v;
}

// Low 32 bits of the generic address of a __shared__ object = LDS byte offset.
__device__ __forceinline__ unsigned int lds_addr_of(const void* p) {
  return (unsigned int)(unsigned long long)p;
}

// ---------------------------------------------------------------------------
// TDM: DMA a rows x 32 bf16 tile (row stride = stride_elems, in elements)
// from global memory into LDS at lds_addr, packed contiguously (row i at
// lds_addr + i*64 bytes).  D# per CDNA5 ISA sec 8.3/8.4 (2-D tensor, groups
// 2/3 zero).  Issued per-wave; tracked with TENSORcnt.
// ---------------------------------------------------------------------------
__device__ __forceinline__ void tdm_load_tile_bf16(unsigned int lds_addr,
                                                   const bf16_t* gptr,
                                                   int rows, int stride_elems) {
  unsigned long long ga = (unsigned long long)gptr;
  u32x4 g0;
  g0[0] = 1u;                                    // count=1 (valid user descriptor)
  g0[1] = lds_addr;                              // lds_addr[31:0]
  g0[2] = (unsigned int)ga;                      // global_addr[31:0]
  g0[3] = (unsigned int)((ga >> 32) & 0x1FFFFFFu) | 0x80000000u;  // addr[56:32] | type=2
  i32x8 g1;
  g1[0] = 0x00010000;                            // data_size=1 (2 bytes), no multicast
  g1[1] = (int)(((unsigned int)stride_elems & 0xFFFFu) << 16);    // tensor_dim0[15:0]
  g1[2] = (int)((((unsigned int)stride_elems >> 16) & 0xFFFFu) |  // tensor_dim0[31:16]
                (((unsigned int)rows & 0xFFFFu) << 16));          // tensor_dim1[15:0]
  g1[3] = (int)(32u << 16);                      // tensor_dim1[31:16]=0 | tile_dim0=32
  g1[4] = (int)((unsigned int)rows & 0xFFFFu);   // tile_dim1=rows, tile_dim2=0
  g1[5] = (int)(unsigned int)stride_elems;       // tensor_dim0_stride[31:0]
  g1[6] = 0;                                     // stride0[47:32]=0, dim1_stride lo=0
  g1[7] = 0;
  i32x4 z4;
#pragma unroll
  for (int i = 0; i < 4; ++i) z4[i] = 0;
#if __has_include(<hip/amd_detail/amd_gfx1250_TDM.h>)
  i32x8 z8;
#pragma unroll
  for (int i = 0; i < 8; ++i) z8[i] = 0;
  __builtin_amdgcn_tensor_load_to_lds(g0, g1, z4, z4, z8, 0);
#else
  __builtin_amdgcn_tensor_load_to_lds(g0, g1, z4, z4, 0);
#endif
}

// Load a 16x32 bf16 fragment (A: rows=M, or B: rows=N of W^T) from an LDS tile
// laid out row-major [16+][32] bf16.  Per ISA 7.12.2 (16-bit A 16x32):
//   lanes 0-15  : row = lane,    K = 0..7 (v0-3) and 16..23 (v4-7)
//   lanes 16-31 : row = lane-16, K = 8..15 and 24..31
// Each 8-element run is 16B contiguous -> ds_load_b128 pairs.
__device__ __forceinline__ v16bf load_frag(const bf16_t* p, int ld) {
  const int lane = threadIdx.x & 31;
  const int r  = lane & 15;
  const int kb = (lane >> 4) << 3;  // 0 or 8
  const bf16_t* row = p + r * ld + kb;
  v16bf f;
#pragma unroll
  for (int i = 0; i < 8; ++i) { f[i] = row[i]; f[8 + i] = row[16 + i]; }
  return f;
}

// Butterfly sum over the 16-lane half-wave (C/D fragment rows live in a half).
__device__ __forceinline__ float half_reduce16(float v) {
#pragma unroll
  for (int off = 1; off < 16; off <<= 1) v += __shfl_xor(v, off, 32);
  return v;
}

__device__ __forceinline__ float gelu_exact(float x) {
  return 0.5f * x * (1.0f + erff(x * 0.70710678118654752f));
}

// ---------------------------------------------------------------------------
// Shared bf16 GEMM inner loop with TDM double buffering.
// 128 rows x (NT*16) cols, K = 32*KSTEPS.
// A : bf16 [128][lda]     (base offset to block row 0, col 0)
// Wt: bf16 [NT*16][ldw]   (W transposed, n-major; base offset to block col 0)
// As: LDS, 2 * 128*32 bf16.   Bs: LDS, 2 * NT*16*32 bf16.
// Wave 0 issues the DMA for step kt+1 while all waves run WMMA on step kt.
// ---------------------------------------------------------------------------
template <int KSTEPS, int NT>
__device__ __forceinline__ void gemm_bf16_tdm(const bf16_t* __restrict__ A, int lda,
                                              const bf16_t* __restrict__ Wt, int ldw,
                                              bf16_t* As, bf16_t* Bs, v8f* acc) {
  const int tid  = threadIdx.x;
  const int wave = tid >> 5;
  const int ABUF = 128 * 32;
  const int BBUF = NT * 16 * 32;
  if (wave == 0) {
    tdm_load_tile_bf16(lds_addr_of(As), A, 128, lda);
    tdm_load_tile_bf16(lds_addr_of(Bs), Wt, NT * 16, ldw);
  }
  int cur = 0;
  for (int kt = 0; kt < KSTEPS; ++kt) {
    if (wave == 0) __builtin_amdgcn_s_wait_tensorcnt(0);  // current tiles resident
    __syncthreads();
    if (wave == 0 && kt + 1 < KSTEPS) {                   // DMA next tiles into other buffer
      tdm_load_tile_bf16(lds_addr_of(As + (cur ^ 1) * ABUF), A + (kt + 1) * 32, 128, lda);
      tdm_load_tile_bf16(lds_addr_of(Bs + (cur ^ 1) * BBUF), Wt + (kt + 1) * 32, NT * 16, ldw);
    }
    v16bf af = load_frag(As + cur * ABUF + wave * 16 * 32, 32);
#pragma unroll
    for (int nt = 0; nt < NT; ++nt) {
      v16bf bf = load_frag(Bs + cur * BBUF + nt * 16 * 32, 32);
      acc[nt] = WMMA_BF16(af, bf, acc[nt]);
    }
    __syncthreads();
    cur ^= 1;
  }
}

// ---------------------------------------------------------------------------
// Prep kernels
// ---------------------------------------------------------------------------
__global__ __launch_bounds__(256) void k_transpose(const float* __restrict__ src,
                                                   bf16_t* __restrict__ dst, int K, int N) {
  int idx = blockIdx.x * 256 + threadIdx.x;
  if (idx < K * N) {
    int k = idx / N, n = idx % N;
    dst[(size_t)n * K + k] = (bf16_t)src[idx];
  }
}

__global__ __launch_bounds__(256) void k_negexp(const float* __restrict__ src,
                                                float* __restrict__ dst, int n) {
  int i = blockIdx.x * 256 + threadIdx.x;
  if (i < n) dst[i] = -__expf(src[i]);
}

__global__ __launch_bounds__(128) void k_meta(const int* __restrict__ site, const int* __restrict__ hour,
                                              const float* __restrict__ se, const float* __restrict__ he,
                                              const float* __restrict__ mw, const float* __restrict__ mb,
                                              float* __restrict__ meta) {
  int b = blockIdx.x, c = threadIdx.x;
  int sid = min(max(site[b], 0), 19);
  int hr  = min(max(hour[b], 0), 23);
  float acc = mb[c];
#pragma unroll
  for (int j = 0; j < 16; ++j)
    acc += se[sid * 16 + j] * mw[j * 128 + c] + he[hr * 16 + j] * mw[(16 + j) * 128 + c];
  meta[(size_t)b * 128 + c] = acc;
}

// ---------------------------------------------------------------------------
// K1: masked input projection  h = gelu(LN(x @ in_W + b)) + pos + meta
// A tile: fp32 emb, masked + converted to bf16 manually (DMA can't convert).
// B tile: bf16 weights via TDM, double-buffered.
// ---------------------------------------------------------------------------
__global__ __launch_bounds__(256) void k_inproj(const float* __restrict__ emb,
                                                const unsigned char* __restrict__ mask,
                                                const bf16_t* __restrict__ wt,   // [128][1536]
                                                const float* __restrict__ in_b,
                                                const float* __restrict__ in_g,
                                                const float* __restrict__ in_beta,
                                                const float* __restrict__ pos,   // [12][128]
                                                const float* __restrict__ meta,  // [B][128]
                                                float* __restrict__ h32, bf16_t* __restrict__ hbf) {
  __shared__ bf16_t As[128 * 32];
  __shared__ bf16_t Bs[2 * 128 * 32];
  const int tid = threadIdx.x, wave = tid >> 5, lane = tid & 31;
  const int rowBase = blockIdx.x * 128;
  const int lr = tid >> 1, c0 = (tid & 1) * 16;
  const int arow = rowBase + lr;
  const float msk = mask[arow] ? 0.0f : 1.0f;
  const float* aptr = emb + (size_t)arow * D_IN + c0;
  const int BBUF = 128 * 32;
  v8f acc[8];
#pragma unroll
  for (int i = 0; i < 8; ++i) acc[i] = zero_v8f();

  if (wave == 0) tdm_load_tile_bf16(lds_addr_of(Bs), wt, 128, D_IN);
  int cur = 0;
  for (int kt = 0; kt < D_IN / 32; ++kt) {
#pragma unroll
    for (int i = 0; i < 16; ++i) As[lr * 32 + c0 + i] = (bf16_t)(aptr[kt * 32 + i] * msk);
    if (kt + 1 < D_IN / 32) __builtin_prefetch(aptr + kt * 32 + 32, 0, 0);  // global_prefetch_b8
    if (wave == 0) __builtin_amdgcn_s_wait_tensorcnt(0);
    __syncthreads();
    if (wave == 0 && kt + 1 < D_IN / 32)
      tdm_load_tile_bf16(lds_addr_of(Bs + (cur ^ 1) * BBUF), wt + (kt + 1) * 32, 128, D_IN);
    v16bf af = load_frag(As + wave * 16 * 32, 32);
#pragma unroll
    for (int nt = 0; nt < 8; ++nt) {
      v16bf bf = load_frag(Bs + cur * BBUF + nt * 16 * 32, 32);
      acc[nt] = WMMA_BF16(af, bf, acc[nt]);
    }
    __syncthreads();
    cur ^= 1;
  }

  // Epilogue: LN over 128 cols via half-wave shuffles, gelu, + pos + meta.
  const int mhi = (lane >> 4) << 3, ln = lane & 15;
#pragma unroll
  for (int j = 0; j < 8; ++j) {
    float vals[8], s = 0.f, sq = 0.f;
#pragma unroll
    for (int nt = 0; nt < 8; ++nt) {
      float v = acc[nt][j] + in_b[nt * 16 + ln];
      vals[nt] = v; s += v; sq += v * v;
    }
    s  = half_reduce16(s);
    sq = half_reduce16(sq);
    float mean = s * (1.0f / 128.0f);
    float var  = sq * (1.0f / 128.0f) - mean * mean;
    float rstd = rsqrtf(var + 1e-5f);
    int row = rowBase + wave * 16 + mhi + j;
    int t = row % T_SEQ, b = row / T_SEQ;
#pragma unroll
    for (int nt = 0; nt < 8; ++nt) {
      int col = nt * 16 + ln;
      float v = (vals[nt] - mean) * rstd * in_g[col] + in_beta[col];
      v = gelu_exact(v);
      v += pos[t * 128 + col] + meta[(size_t)b * 128 + col];
      h32[(size_t)row * 128 + col] = v;
      hbf[(size_t)row * 128 + col] = (bf16_t)v;
    }
  }
}

// ---------------------------------------------------------------------------
// K2: fused delta/B/C projection: h @ [W_delta|W_B|W_C]  (K=128, N=160)
// proj layout per row: [0..127]=softplus(delta), [128..143]=Bm, [144..159]=Cm
// ---------------------------------------------------------------------------
__global__ __launch_bounds__(256) void k_proj(const bf16_t* __restrict__ hbf,
                                              const bf16_t* __restrict__ wcat,  // [160][128]
                                              const float* __restrict__ b_delta,
                                              const float* __restrict__ b_B,
                                              const float* __restrict__ b_C,
                                              float* __restrict__ proj) {
  __shared__ bf16_t As[2 * 128 * 32];
  __shared__ bf16_t Bs[2 * 160 * 32];
  const int rowBase = blockIdx.x * 128;
  v8f acc[10];
#pragma unroll
  for (int i = 0; i < 10; ++i) acc[i] = zero_v8f();
  gemm_bf16_tdm<4, 10>(hbf + (size_t)rowBase * 128, 128, wcat, 128, As, Bs, acc);

  const int wave = threadIdx.x >> 5, lane = threadIdx.x & 31;
  const int mhi = (lane >> 4) << 3, ln = lane & 15;
#pragma unroll
  for (int nt = 0; nt < 10; ++nt) {
    int col = nt * 16 + ln;
    float bias = (col < 128) ? b_delta[col] : (col < 144 ? b_B[col - 128] : b_C[col - 144]);
#pragma unroll
    for (int j = 0; j < 8; ++j) {
      int row = rowBase + wave * 16 + mhi + j;
      float v = acc[nt][j] + bias;
      if (col < 128) v = (v > 20.0f) ? v : log1pf(__expf(v));  // softplus
      proj[(size_t)row * 160 + col] = v;
    }
  }
}

// ---------------------------------------------------------------------------
// K3: selective scan.  One block per batch element, one thread per channel d.
// reverse=1 runs the time loop backwards (bwd SSM on reversed sequence,
// outputs already land at the original positions).
// ---------------------------------------------------------------------------
__global__ __launch_bounds__(128) void k_scan(const float* __restrict__ proj,  // [BT][160]
                                              const float* __restrict__ h32,   // [BT][128] layer input x
                                              const float* __restrict__ Aneg,  // [128][16]
                                              const float* __restrict__ Dvec,  // [128]
                                              bf16_t* __restrict__ hcat,       // [BT][256]
                                              int colOff, int reverse) {
  __shared__ float Bsm[T_SEQ][16];
  __shared__ float Csm[T_SEQ][16];
  const int b = blockIdx.x, d = threadIdx.x;
  for (int i = d; i < T_SEQ * 16; i += 128) {
    int t = i >> 4, n = i & 15;
    const float* pr = proj + ((size_t)b * T_SEQ + t) * 160;
    Bsm[t][n] = pr[128 + n];
    Csm[t][n] = pr[144 + n];
  }
  __syncthreads();
  float Ar[16], s[16];
#pragma unroll
  for (int n = 0; n < 16; ++n) { Ar[n] = Aneg[d * 16 + n]; s[n] = 0.0f; }
  const float Dd = Dvec[d];
  for (int ti = 0; ti < T_SEQ; ++ti) {
    int t = reverse ? (T_SEQ - 1 - ti) : ti;
    size_t row = (size_t)b * T_SEQ + t;
    float delta = proj[row * 160 + d];
    float x     = h32[row * 128 + d];
    float dx    = delta * x;
    float y = 0.0f;
#pragma unroll
    for (int n = 0; n < 16; ++n) {
      float dA = __expf(delta * Ar[n]);
      s[n] = dA * s[n] + dx * Bsm[t][n];
      y += s[n] * Csm[t][n];
    }
    hcat[row * 256 + colOff + d] = (bf16_t)(y + x * Dd);
  }
}

// ---------------------------------------------------------------------------
// K4: merge = concat(h_f,h_b) @ merge_W + b, + residual, LayerNorm.
// ---------------------------------------------------------------------------
__global__ __launch_bounds__(256) void k_merge(const bf16_t* __restrict__ hcat,
                                               const bf16_t* __restrict__ wt,  // [128][256]
                                               const float* __restrict__ mb,
                                               const float* __restrict__ lng,
                                               const float* __restrict__ lnb,
                                               float* __restrict__ h32, bf16_t* __restrict__ hbf) {
  __shared__ bf16_t As[2 * 128 * 32];
  __shared__ bf16_t Bs[2 * 128 * 32];
  const int rowBase = blockIdx.x * 128;
  v8f acc[8];
#pragma unroll
  for (int i = 0; i < 8; ++i) acc[i] = zero_v8f();
  gemm_bf16_tdm<8, 8>(hcat + (size_t)rowBase * 256, 256, wt, 256, As, Bs, acc);

  const int wave = threadIdx.x >> 5, lane = threadIdx.x & 31;
  const int mhi = (lane >> 4) << 3, ln = lane & 15;
#pragma unroll
  for (int j = 0; j < 8; ++j) {
    int row = rowBase + wave * 16 + mhi + j;
    float vals[8], s = 0.f, sq = 0.f;
#pragma unroll
    for (int nt = 0; nt < 8; ++nt) {
      int col = nt * 16 + ln;
      float v = acc[nt][j] + mb[col] + h32[(size_t)row * 128 + col];  // + residual
      vals[nt] = v; s += v; sq += v * v;
    }
    s  = half_reduce16(s);
    sq = half_reduce16(sq);
    float mean = s * (1.0f / 128.0f);
    float var  = sq * (1.0f / 128.0f) - mean * mean;
    float rstd = rsqrtf(var + 1e-5f);
#pragma unroll
    for (int nt = 0; nt < 8; ++nt) {
      int col = nt * 16 + ln;
      float v = (vals[nt] - mean) * rstd * lng[col] + lnb[col];
      h32[(size_t)row * 128 + col] = v;
      hbf[(size_t)row * 128 + col] = (bf16_t)v;
    }
  }
}

// ---------------------------------------------------------------------------
// K5: rec hidden = gelu(h @ rec_W1 + b1)   (K=128, N=256 -> gridDim.y = 2)
// ---------------------------------------------------------------------------
__global__ __launch_bounds__(256) void k_rec1(const bf16_t* __restrict__ hbf,
                                              const bf16_t* __restrict__ wt,  // [256][128]
                                              const float* __restrict__ b1,
                                              bf16_t* __restrict__ gbf) {
  __shared__ bf16_t As[2 * 128 * 32];
  __shared__ bf16_t Bs[2 * 128 * 32];
  const int rowBase = blockIdx.x * 128;
  const int colBase = blockIdx.y * 128;
  v8f acc[8];
#pragma unroll
  for (int i = 0; i < 8; ++i) acc[i] = zero_v8f();
  gemm_bf16_tdm<4, 8>(hbf + (size_t)rowBase * 128, 128, wt + (size_t)colBase * 128, 128, As, Bs, acc);

  const int wave = threadIdx.x >> 5, lane = threadIdx.x & 31;
  const int mhi = (lane >> 4) << 3, ln = lane & 15;
#pragma unroll
  for (int nt = 0; nt < 8; ++nt) {
    int col = colBase + nt * 16 + ln;
    float bias = b1[col];
#pragma unroll
    for (int j = 0; j < 8; ++j) {
      int row = rowBase + wave * 16 + mhi + j;
      gbf[(size_t)row * 256 + col] = (bf16_t)gelu_exact(acc[nt][j] + bias);
    }
  }
}

// ---------------------------------------------------------------------------
// K6: pred = g @ rec_W2 + b2   (K=256, N=1536 -> gridDim.y = 12)
// ---------------------------------------------------------------------------
__global__ __launch_bounds__(256) void k_rec2(const bf16_t* __restrict__ gbf,
                                              const bf16_t* __restrict__ wt,  // [1536][256]
                                              const float* __restrict__ b2,
                                              float* __restrict__ out) {
  __shared__ bf16_t As[2 * 128 * 32];
  __shared__ bf16_t Bs[2 * 128 * 32];
  const int rowBase = blockIdx.x * 128;
  const int colBase = blockIdx.y * 128;
  v8f acc[8];
#pragma unroll
  for (int i = 0; i < 8; ++i) acc[i] = zero_v8f();
  gemm_bf16_tdm<8, 8>(gbf + (size_t)rowBase * 256, 256, wt + (size_t)colBase * 256, 256, As, Bs, acc);

  const int wave = threadIdx.x >> 5, lane = threadIdx.x & 31;
  const int mhi = (lane >> 4) << 3, ln = lane & 15;
#pragma unroll
  for (int nt = 0; nt < 8; ++nt) {
    int col = colBase + nt * 16 + ln;
    float bias = b2[col];
#pragma unroll
    for (int j = 0; j < 8; ++j) {
      int row = rowBase + wave * 16 + mhi + j;
      out[(size_t)row * D_IN + col] = acc[nt][j] + bias;
    }
  }
}

// ---------------------------------------------------------------------------
// Host-side orchestration.
// Input pointer order assumes recursive dict-insertion-order flattening:
//  0 emb, 1 mask(bool/u8), 2 site_ids, 3 hours, then params:
//  4 in_W, 5 in_b, 6 in_g, 7 in_beta, 8 pos, 9 site_emb, 10 hour_emb,
//  11 meta_W, 12 meta_b,
//  per layer (x2): fwd{A_log,D,W_delta,b_delta,W_B,b_B,W_C,b_C},
//                  bwd{...same 8...}, merge_W, merge_b, ln_g, ln_b,
//  then rec_W1, rec_b1, rec_W2, rec_b2.                (n_in = 57)
// ---------------------------------------------------------------------------
extern "C" void kernel_launch(void* const* d_in, const int* in_sizes, int n_in,
                              void* d_out, int out_size, void* d_ws, size_t ws_size,
                              hipStream_t stream) {
  (void)in_sizes; (void)n_in; (void)out_size; (void)ws_size;

  const float* emb             = (const float*)d_in[0];
  const unsigned char* mask    = (const unsigned char*)d_in[1];
  const int* site_ids          = (const int*)d_in[2];
  const int* hours             = (const int*)d_in[3];
  auto F = [&](int i) { return (const float*)d_in[i]; };
  const float* in_W = F(4);  const float* in_b = F(5);
  const float* in_g = F(6);  const float* in_beta = F(7);
  const float* pos = F(8);   const float* site_emb = F(9);
  const float* hour_emb = F(10);
  const float* meta_W = F(11); const float* meta_b = F(12);

  struct SSM { const float *A_log, *D, *W_delta, *b_delta, *W_B, *b_B, *W_C, *b_C; };
  struct Layer { SSM fwd, bwd; const float *merge_W, *merge_b, *ln_g, *ln_b; };
  Layer L[2];
  int idx = 13;
  for (int l = 0; l < 2; ++l) {
    SSM* ss[2] = { &L[l].fwd, &L[l].bwd };
    for (int d = 0; d < 2; ++d) {
      ss[d]->A_log = F(idx++); ss[d]->D = F(idx++);
      ss[d]->W_delta = F(idx++); ss[d]->b_delta = F(idx++);
      ss[d]->W_B = F(idx++); ss[d]->b_B = F(idx++);
      ss[d]->W_C = F(idx++); ss[d]->b_C = F(idx++);
    }
    L[l].merge_W = F(idx++); L[l].merge_b = F(idx++);
    L[l].ln_g = F(idx++);    L[l].ln_b = F(idx++);
  }
  const float* rec_W1 = F(idx++); const float* rec_b1 = F(idx++);
  const float* rec_W2 = F(idx++); const float* rec_b2 = F(idx++);

  // ---- workspace layout (~78 MB) ----
  char* wsb = (char*)d_ws;
  size_t off = 0;
  auto walloc = [&](size_t bytes) -> void* {
    void* p = wsb + off;
    off = (off + bytes + 255) & ~(size_t)255;
    return p;
  };
  bf16_t* WT_IN = (bf16_t*)walloc((size_t)128 * 1536 * 2);    // in_W^T
  bf16_t* WCAT  = (bf16_t*)walloc((size_t)4 * 160 * 128 * 2); // [Wd|WB|WC]^T per (layer,dir)
  bf16_t* WMRG  = (bf16_t*)walloc((size_t)2 * 128 * 256 * 2); // merge_W^T per layer
  bf16_t* WR1   = (bf16_t*)walloc((size_t)256 * 128 * 2);     // rec_W1^T
  bf16_t* WR2   = (bf16_t*)walloc((size_t)1536 * 256 * 2);    // rec_W2^T
  float*  A4    = (float*)walloc((size_t)4 * 2048 * 4);       // -exp(A_log) per (layer,dir)
  float*  METAV = (float*)walloc((size_t)BATCH * 128 * 4);
  float*  H32   = (float*)walloc((size_t)BT * 128 * 4);
  bf16_t* HBF   = (bf16_t*)walloc((size_t)BT * 128 * 2);
  float*  PROJF = (float*)walloc((size_t)BT * 160 * 4);
  float*  PROJB = (float*)walloc((size_t)BT * 160 * 4);
  bf16_t* HCAT  = (bf16_t*)walloc((size_t)BT * 256 * 2);
  bf16_t* GBF   = (bf16_t*)walloc((size_t)BT * 256 * 2);

  auto T256 = [](int n) { return (n + 255) / 256; };

  // ---- weight prep ----
  k_transpose<<<T256(1536 * 128), 256, 0, stream>>>(in_W, WT_IN, 1536, 128);
  for (int l = 0; l < 2; ++l) {
    const SSM* ss[2] = { &L[l].fwd, &L[l].bwd };
    for (int d = 0; d < 2; ++d) {
      bf16_t* wc = WCAT + (size_t)(l * 2 + d) * 160 * 128;
      k_transpose<<<T256(128 * 128), 256, 0, stream>>>(ss[d]->W_delta, wc, 128, 128);
      k_transpose<<<T256(128 * 16), 256, 0, stream>>>(ss[d]->W_B, wc + 128 * 128, 128, 16);
      k_transpose<<<T256(128 * 16), 256, 0, stream>>>(ss[d]->W_C, wc + 144 * 128, 128, 16);
      k_negexp<<<T256(2048), 256, 0, stream>>>(ss[d]->A_log, A4 + (size_t)(l * 2 + d) * 2048, 2048);
    }
    k_transpose<<<T256(256 * 128), 256, 0, stream>>>(L[l].merge_W, WMRG + (size_t)l * 128 * 256, 256, 128);
  }
  k_transpose<<<T256(128 * 256), 256, 0, stream>>>(rec_W1, WR1, 128, 256);
  k_transpose<<<T256(256 * 1536), 256, 0, stream>>>(rec_W2, WR2, 256, 1536);
  k_meta<<<BATCH, 128, 0, stream>>>(site_ids, hours, site_emb, hour_emb, meta_W, meta_b, METAV);

  // ---- encoder ----
  k_inproj<<<BT / 128, 256, 0, stream>>>(emb, mask, WT_IN, in_b, in_g, in_beta, pos, METAV, H32, HBF);

  for (int l = 0; l < 2; ++l) {
    const SSM* ss[2] = { &L[l].fwd, &L[l].bwd };
    float* proj[2] = { PROJF, PROJB };
    for (int d = 0; d < 2; ++d) {
      k_proj<<<BT / 128, 256, 0, stream>>>(HBF, WCAT + (size_t)(l * 2 + d) * 160 * 128,
                                           ss[d]->b_delta, ss[d]->b_B, ss[d]->b_C, proj[d]);
    }
    for (int d = 0; d < 2; ++d) {
      k_scan<<<BATCH, 128, 0, stream>>>(proj[d], H32, A4 + (size_t)(l * 2 + d) * 2048,
                                        ss[d]->D, HCAT, d * 128, d);
    }
    k_merge<<<BT / 128, 256, 0, stream>>>(HCAT, WMRG + (size_t)l * 128 * 256,
                                          L[l].merge_b, L[l].ln_g, L[l].ln_b, H32, HBF);
  }

  // ---- reconstruction head ----
  k_rec1<<<dim3(BT / 128, 2), 256, 0, stream>>>(HBF, WR1, rec_b1, GBF);
  k_rec2<<<dim3(BT / 128, 12), 256, 0, stream>>>(GBF, WR2, rec_b2, (float*)d_out);
}
